// BiRNNModel_31318901522893
// MI455X (gfx1250) — matrix operand order repaired and checked by reading, not compiled, not run on hardware
//
#include <hip/hip_runtime.h>
#include <math.h>

// ---------------------------------------------------------------------------
// BiLSTM (B=64, S=512, H=1024, L=2) + dense head for MI455X (gfx1250).
//  * bf16 recurrent weight set = 64MB -> fully L2-resident (192MB L2); the
//    sequential recurrence never touches HBM after the first pass, so the
//    loop is latency/WMMA bound, not bandwidth bound.
//  * One persistent kernel: 128 WGs x 512 threads (64 WGs / direction).
//    Each WG owns a 16-wide hidden slice, computes all 4 gates for it with
//    v_wmma_f32_16x16x32_bf16 (16 waves = 4 M-tiles x 4 gates), keeps
//    c-state in LDS, ping-pongs h in global/L2, syncs steps with
//    device-scope atomic grid barriers.
//  * A-operand streams into LDS with gfx1250 async copies
//    (global_load_async_to_lds_b128 + s_wait_asynccnt), double-buffered in
//    K=128 stages; per stage all 8 B-frag global b128 + 8 A-frag ds b128
//    loads are batched ahead of the 4 WMMAs for deep MLP.
// ---------------------------------------------------------------------------

#define BB 64
#define SS 512
#define HH 1024
#define LL 2
#define NL 1000

typedef __attribute__((ext_vector_type(16))) __bf16 v16bf;
typedef __attribute__((ext_vector_type(8)))  float  v8f;
typedef __attribute__((ext_vector_type(8)))  unsigned int v8u;

__device__ __forceinline__ unsigned short f2bf(float x) {
  unsigned int u = __builtin_bit_cast(unsigned int, x);
  unsigned int r = u + 0x7FFFu + ((u >> 16) & 1u);
  return (unsigned short)(r >> 16);
}

// A-fragment K-offset within a 32-wide K chunk (16-bit A layout, ISA 7.12.2)
__device__ __forceinline__ int a_kbase(int v, int kh) {
  return (v < 4) ? (kh * 8 + 2 * v) : (16 + kh * 8 + 2 * (v - 4));
}

// gfx1250 async global->LDS copy, 16B per lane (ASYNCcnt-tracked).
// LDS address = low 32 bits of the generic pointer (flat LDS = aperture|offset).
__device__ __forceinline__ void async_copy_b128(const unsigned short* g,
                                                const unsigned short* l) {
  unsigned lds = (unsigned)(uintptr_t)(const void*)l;
  unsigned long long ga = (unsigned long long)(uintptr_t)(const void*)g;
  asm volatile("global_load_async_to_lds_b128 %0, %1, off"
               :: "v"(lds), "v"(ga) : "memory");
}
__device__ __forceinline__ void wait_async_le2() {
  asm volatile("s_wait_asynccnt 0x2" ::: "memory");
}
__device__ __forceinline__ void wait_async_0() {
  asm volatile("s_wait_asynccnt 0x0" ::: "memory");
}

// ---------------------------------------------------------------------------
// Packing / init kernels
// ---------------------------------------------------------------------------

// W [nLayers][1024][4096] f32 -> packed bf16 B-fragments:
// block = (nt, kt) of a 32x16 (KxN) tile; lane L holds n = nt*16+(L&15),
// K = kt*32 + (L>>4)*16 + q (q=0..15 contiguous) -> 2 coalesced b128/lane.
__global__ void pack_w_kernel(const float* __restrict__ W,
                              unsigned short* __restrict__ out, int nLayers) {
  int t = blockIdx.x * blockDim.x + threadIdx.x;
  int total = nLayers * 256 * 32 * 32;
  if (t >= total) return;
  int lane = t & 31;
  int kt = (t >> 5) & 31;
  int nt = (t >> 10) & 255;
  int l  = t >> 18;
  int n  = nt * 16 + (lane & 15);
  int kh = lane >> 4;
  const float* w = W + (size_t)l * HH * 4 * HH;
  unsigned short* o = out + ((size_t)(l * 256 * 32 + nt * 32 + kt) * 32 + lane) * 16;
#pragma unroll
  for (int q = 0; q < 16; ++q) {
    int k = kt * 32 + kh * 16 + q;
    o[q] = f2bf(w[(size_t)k * (4 * HH) + n]);
  }
}

// Wd [2048][1000] -> packed [63][64] blocks, zero-padded to 1008 columns.
__global__ void pack_wd_kernel(const float* __restrict__ Wd,
                               unsigned short* __restrict__ out) {
  int t = blockIdx.x * blockDim.x + threadIdx.x;
  int total = 63 * 64 * 32;
  if (t >= total) return;
  int lane = t & 31;
  int kt = (t >> 5) & 63;
  int nt = t >> 11;
  int n  = nt * 16 + (lane & 15);
  int kh = lane >> 4;
  unsigned short* o = out + ((size_t)(nt * 64 + kt) * 32 + lane) * 16;
#pragma unroll
  for (int q = 0; q < 16; ++q) {
    int k = kt * 32 + kh * 16 + q;
    o[q] = (n < NL) ? f2bf(Wd[(size_t)k * NL + n]) : (unsigned short)0;
  }
}

// Embedding gather to time-major bf16 x[S][B][H].
__global__ void gather_x_kernel(const int* __restrict__ ids,
                                const float* __restrict__ emb,
                                unsigned short* __restrict__ x_tm) {
  long t = (long)blockIdx.x * blockDim.x + threadIdx.x;  // S*B*(H/2)
  if (t >= (long)SS * BB * (HH / 2)) return;
  int hh = (int)(t & (HH / 2 - 1)) * 2;
  int b  = (int)(t >> 9) & (BB - 1);
  int s  = (int)(t >> 15);
  int row = ids[b * SS + s];
  const float* e = emb + (long)row * HH + hh;
  unsigned short* o = x_tm + ((long)s * BB + b) * HH + hh;
  o[0] = f2bf(e[0]);
  o[1] = f2bf(e[1]);
}

// Zero h-state + barrier counters.
__global__ void init_state_kernel(unsigned int* __restrict__ h,
                                  unsigned int* __restrict__ cnt, int nWords) {
  int t = blockIdx.x * blockDim.x + threadIdx.x;
  if (t < nWords) h[t] = 0u;
  if (t == 0) { cnt[0] = 0u; cnt[1] = 0u; }
}

// ---------------------------------------------------------------------------
// Persistent bidirectional recurrent kernel
// ---------------------------------------------------------------------------
__device__ __forceinline__ void grid_barrier(unsigned int* cnt, unsigned int target) {
  __threadfence();                       // release h writes to device scope
  __syncthreads();
  if (threadIdx.x == 0) {
    atomicAdd(cnt, 1u);
    while (__hip_atomic_load(cnt, __ATOMIC_RELAXED, __HIP_MEMORY_SCOPE_AGENT) < target) {
      __builtin_amdgcn_s_sleep(2);
    }
  }
  __syncthreads();
  __threadfence();                       // acquire other WGs' h writes
}

__launch_bounds__(512, 1)
__global__ void birnn_recurrent_kernel(
    const unsigned short* __restrict__ x_tm,   // [S][B][H] bf16
    const unsigned short* __restrict__ wx_f,   // packed [L][256*32*512]
    const unsigned short* __restrict__ wh_f,
    const unsigned short* __restrict__ wx_b,
    const unsigned short* __restrict__ wh_b,
    const float* __restrict__ b_f,             // [L][4H]
    const float* __restrict__ b_b,
    unsigned short* __restrict__ h_state,      // [dir][L][parity][B][H] bf16
    unsigned int* __restrict__ barrier_cnt)    // [2]
{
  const int tid  = threadIdx.x;
  const int dir  = blockIdx.x >> 6;            // 64 WGs per direction
  const int wg   = blockIdx.x & 63;
  const int j0   = wg * 16;                    // this WG's hidden slice
  const int wave = tid >> 5;
  const int lane = tid & 31;
  const int gate = wave >> 2;                  // i,f,g,o
  const int mt   = wave & 3;                   // M tile (rows mt*16..)
  const int lm   = lane & 15;
  const int kh   = lane >> 4;
  const int m_row = mt * 16 + lm;              // A-fragment row for this lane

  const unsigned short* Wx = dir ? wx_b : wx_f;
  const unsigned short* Wh = dir ? wh_b : wh_f;
  const float* bias = dir ? b_b : b_f;
  unsigned short* hbase = h_state + (size_t)dir * (LL * 2 * BB * HH);
  unsigned int* cnt = barrier_cnt + dir;

  __shared__ unsigned short ablk[2][BB * 128]; // 2x16KB double-buffered A stage
  __shared__ float gateBuf[4 * BB * 16];       // 16KB gate exchange
  __shared__ float cState[LL][BB * 16];        // 8KB cell state (WG-local)

  for (int e = tid; e < BB * 16; e += 512) { cState[0][e] = 0.f; cState[1][e] = 0.f; }
  __syncthreads();

  // per-thread transfer indices for a 16KB stage (1024 x 16B, 2 per thread)
  const int i0_row = tid >> 4, i0_seg = tid & 15;
  const int i1 = tid + 512;
  const int i1_row = i1 >> 4, i1_seg = i1 & 15;

  const int ntg = gate * 64 + wg;              // global N tile 0..255
  unsigned int phase = 0;

  for (int t = 0; t < SS; ++t) {
    const int p  = t & 1;
    const int wp = p ^ 1;
    const int s  = dir ? (SS - 1 - t) : t;
    const unsigned short* x_t = x_tm + (size_t)s * BB * HH;

    for (int l = 0; l < LL; ++l) {
      const unsigned short* inp   = (l == 0) ? x_t : (hbase + (size_t)(0 * 2 + wp) * BB * HH);
      const unsigned short* hprev = hbase + (size_t)(l * 2 + p) * BB * HH;

      const float bv = bias[l * 4 * HH + gate * HH + j0 + lm];
      v8f acc = {bv, bv, bv, bv, bv, bv, bv, bv};

      for (int part = 0; part < 2; ++part) {
        const unsigned short* A = part ? hprev : inp;
        const unsigned short* Wp =
            (part ? Wh : Wx) + (size_t)l * (256 * 32 * 512) + (size_t)ntg * 32 * 512 + lane * 16;

        // prologue: async-stage chunk 0 (K = 0..127)
        async_copy_b128(A + (size_t)i0_row * HH + i0_seg * 8, &ablk[0][i0_row * 128 + i0_seg * 8]);
        async_copy_b128(A + (size_t)i1_row * HH + i1_seg * 8, &ablk[0][i1_row * 128 + i1_seg * 8]);

        for (int st = 0; st < 8; ++st) {
          const unsigned short* bufp = ablk[st & 1];
          if (st < 7) {                         // prefetch next stage into other buffer
            unsigned short* nb = ablk[(st + 1) & 1];
            const unsigned short* As = A + (st + 1) * 128;
            async_copy_b128(As + (size_t)i0_row * HH + i0_seg * 8, nb + i0_row * 128 + i0_seg * 8);
            async_copy_b128(As + (size_t)i1_row * HH + i1_seg * 8, nb + i1_row * 128 + i1_seg * 8);
            wait_async_le2();                   // stage st's 2 copies done, next 2 in flight
          } else {
            wait_async_0();
          }
          __syncthreads();                      // all waves' portions of stage st landed

          // batch ALL loads of the stage (8 global b128 + 8 ds b128) for MLP,
          // then run the 4 dependent WMMAs.
          v8u au[4], bu[4];
#pragma unroll
          for (int j = 0; j < 4; ++j) {
            const unsigned short* bp = Wp + (size_t)(st * 4 + j) * 512;
            *((uint4*)&bu[j])     = *(const uint4*)bp;
            *((uint4*)&bu[j] + 1) = *(const uint4*)(bp + 8);
            *((uint4*)&au[j])     = *(const uint4*)(bufp + m_row * 128 + j * 32 + kh * 16);
            *((uint4*)&au[j] + 1) = *(const uint4*)(bufp + m_row * 128 + j * 32 + kh * 16 + 8);
          }
          __builtin_prefetch(Wp + (size_t)(st * 4 + 4) * 512, 0, 1);  // next stage's B
#pragma unroll
          for (int j = 0; j < 4; ++j) {
            v16bf a = __builtin_bit_cast(v16bf, au[j]);
            v16bf b = __builtin_bit_cast(v16bf, bu[j]);
            acc = __builtin_amdgcn_wmma_f32_16x16x32_bf16(
                false, a, false, b, (short)0, acc, false, false);
          }
          __syncthreads();                      // readers done before buffer is overwritten
        }
      }

      // exchange the 4 gate tiles through LDS (C layout: vgpr r -> M=r/r+8, N=lane&15)
#pragma unroll
      for (int r = 0; r < 8; ++r) {
        const int mm = (lane < 16) ? r : (r + 8);
        gateBuf[gate * (BB * 16) + (mt * 16 + mm) * 16 + lm] = acc[r];
      }
      __syncthreads();

      // elementwise LSTM cell update for this WG's 64x16 slice
      unsigned short* hout = hbase + (size_t)(l * 2 + wp) * BB * HH;
      for (int e = tid; e < BB * 16; e += 512) {
        const float gi = gateBuf[0 * (BB * 16) + e];
        const float gf = gateBuf[1 * (BB * 16) + e];
        const float gg = gateBuf[2 * (BB * 16) + e];
        const float go = gateBuf[3 * (BB * 16) + e];
        const float si = 1.f / (1.f + __expf(-gi));
        const float sf = 1.f / (1.f + __expf(-gf));
        const float so = 1.f / (1.f + __expf(-go));
        const float cn = sf * cState[l][e] + si * tanhf(gg);
        cState[l][e] = cn;
        const float hn = so * tanhf(cn);
        const int m = e >> 4, n = e & 15;
        hout[(size_t)m * HH + j0 + n] = f2bf(hn);
      }

      ++phase;
      grid_barrier(cnt, phase * 64u);          // all WGs of this direction
    }
  }
}

// ---------------------------------------------------------------------------
// Dense head: out[64,1000] = concat(h_f, h_b)[64,2048] @ Wd + bd  (WMMA)
// One wave per 16x16 output tile; grid = 4 M-tiles x 63 N-tiles.
// ---------------------------------------------------------------------------
__launch_bounds__(32, 1)
__global__ void dense_out_kernel(const unsigned short* __restrict__ hf,
                                 const unsigned short* __restrict__ hb,
                                 const unsigned short* __restrict__ wd_p,
                                 const float* __restrict__ bd,
                                 float* __restrict__ out) {
  const int mt = blockIdx.x & 3;
  const int nt = blockIdx.x >> 2;              // 0..62
  const int lane = threadIdx.x;
  const int lm = lane & 15, kh = lane >> 4;
  const int n = nt * 16 + lm;
  const int m = mt * 16 + lm;

  const float bv = (n < NL) ? bd[n] : 0.f;
  v8f acc = {bv, bv, bv, bv, bv, bv, bv, bv};

  for (int kt = 0; kt < 64; ++kt) {
    const unsigned short* src = (kt < 32) ? hf : hb;
    const int chunk = (kt & 31) * 32;
    v8u au;
#pragma unroll
    for (int v = 0; v < 8; ++v) {
      au[v] = *(const unsigned int*)(src + (size_t)m * HH + chunk + kh * 16 + 2 * v);
    }
    const unsigned short* bp = wd_p + ((size_t)(nt * 64 + kt) * 32 + lane) * 16;
    v8u bu;
    *((uint4*)&bu)     = *(const uint4*)bp;
    *((uint4*)&bu + 1) = *(const uint4*)(bp + 8);
    v16bf a = __builtin_bit_cast(v16bf, au);
    v16bf b = __builtin_bit_cast(v16bf, bu);
    acc = __builtin_amdgcn_wmma_f32_16x16x32_bf16(false, a, false, b, (short)0, acc,
                                                  false, false);
  }
  if (n < NL) {
#pragma unroll
    for (int r = 0; r < 8; ++r) {
      const int mm = (lane < 16) ? r : (r + 8);
      out[(size_t)(mt * 16 + mm) * NL + n] = acc[r];
    }
  }
}

// ---------------------------------------------------------------------------
// Host launcher
// ---------------------------------------------------------------------------
extern "C" void kernel_launch(void* const* d_in, const int* in_sizes, int n_in,
                              void* d_out, int out_size, void* d_ws, size_t ws_size,
                              hipStream_t stream) {
  (void)in_sizes; (void)n_in; (void)out_size; (void)ws_size;
  const int*   ids  = (const int*)d_in[0];
  const float* emb  = (const float*)d_in[1];
  const float* Wx_f = (const float*)d_in[2];
  const float* Wh_f = (const float*)d_in[3];
  const float* b_f  = (const float*)d_in[4];
  const float* Wx_b = (const float*)d_in[5];
  const float* Wh_b = (const float*)d_in[6];
  const float* b_b  = (const float*)d_in[7];
  const float* Wd   = (const float*)d_in[8];
  const float* bd   = (const float*)d_in[9];
  float* out = (float*)d_out;

  char* ws = (char*)d_ws;
  size_t off = 0;
  auto carve = [&](size_t bytes) -> char* {
    off = (off + 255) & ~(size_t)255;
    char* p = ws + off;
    off += bytes;
    return p;
  };
  const size_t wElems  = (size_t)LL * 256 * 32 * 512;     // per packed Wx/Wh tensor
  unsigned short* wxf_p = (unsigned short*)carve(wElems * 2);
  unsigned short* whf_p = (unsigned short*)carve(wElems * 2);
  unsigned short* wxb_p = (unsigned short*)carve(wElems * 2);
  unsigned short* whb_p = (unsigned short*)carve(wElems * 2);
  unsigned short* wd_p  = (unsigned short*)carve((size_t)63 * 64 * 512 * 2);
  unsigned short* x_tm  = (unsigned short*)carve((size_t)SS * BB * HH * 2);
  unsigned short* h_st  = (unsigned short*)carve((size_t)2 * LL * 2 * BB * HH * 2);
  unsigned int*   cnt   = (unsigned int*)carve(256);

  // 1) init h-state + barrier counters
  {
    const int nWords = 2 * LL * 2 * BB * HH / 2;  // ushorts -> uints
    init_state_kernel<<<(nWords + 255) / 256, 256, 0, stream>>>((unsigned int*)h_st, cnt, nWords);
  }
  // 2) embedding gather -> bf16 time-major
  {
    const long total = (long)SS * BB * (HH / 2);
    gather_x_kernel<<<(unsigned)((total + 255) / 256), 256, 0, stream>>>(ids, emb, x_tm);
  }
  // 3) weight packing (bf16 fragment layout)
  {
    const int total = LL * 256 * 32 * 32;
    const int blks = (total + 255) / 256;
    pack_w_kernel<<<blks, 256, 0, stream>>>(Wx_f, wxf_p, LL);
    pack_w_kernel<<<blks, 256, 0, stream>>>(Wh_f, whf_p, LL);
    pack_w_kernel<<<blks, 256, 0, stream>>>(Wx_b, wxb_p, LL);
    pack_w_kernel<<<blks, 256, 0, stream>>>(Wh_b, whb_p, LL);
    const int totald = 63 * 64 * 32;
    pack_wd_kernel<<<(totald + 255) / 256, 256, 0, stream>>>(Wd, wd_p);
  }
  // 4) persistent bidirectional recurrence (L2-resident weights)
  birnn_recurrent_kernel<<<128, 512, 0, stream>>>(x_tm, wxf_p, whf_p, wxb_p, whb_p,
                                                  b_f, b_b, h_st, cnt);
  // 5) dense head. Final top-layer h lands in parity 0 (S even).
  const unsigned short* hf = h_st + (size_t)(0 * 4 + 2 + 0) * BB * HH;  // dir0,l1,p0
  const unsigned short* hb = h_st + (size_t)(1 * 4 + 2 + 0) * BB * HH;  // dir1,l1,p0
  dense_out_kernel<<<252, 32, 0, stream>>>(hf, hb, wd_p, bd, out);
}
